// MultiHeadedAttentionRelative_35837207118244
// MI455X (gfx1250) — matrix operand order repaired and checked by reading, not compiled, not run on hardware
//
#include <hip/hip_runtime.h>
#include <hip/hip_bf16.h>
#include <math.h>

// ---------------------------------------------------------------------------
// Relative-position multi-head attention, CDNA5 (gfx1250), wave32 WMMA.
// B=8, L=1024, D=1024, H=16, DH=64, MRP=16, VOCAB=33.
//
//   k0a: fp32 activations -> f16 (once; halves GEMM A-side traffic)
//   k0b: weight matrices fp32[k][n] -> f16 [n][k]
//   k1 : unified WMMA GEMM, 64x64 tile per wave (4x A/B fragment reuse),
//        modes: QK layout / V-transposed layout / fp32 plain (+bias, scale)
//   k2 : fused attention per (b,h,16-row q tile); 16x1024 score strip in LDS
//        (keeps the 512 MB fp32 logits tensor off HBM); relative-logit via 33-entry
//        per-row table; relative-value via bucket sums (softmax rows sum to 1)
//   k3 : output projection = k1 in mode 2
// ---------------------------------------------------------------------------

typedef __attribute__((ext_vector_type(16))) _Float16 v16h;
typedef __attribute__((ext_vector_type(8)))  float    v8f;
typedef __attribute__((ext_vector_type(4)))  int      v4i;

#define Bq 8
#define Lq 1024
#define Dq 1024
#define Hq 16
#define DHq 64
#define VOCABq 33

#if defined(__has_builtin)
#if __has_builtin(__builtin_amdgcn_global_load_async_to_lds_b128) && \
    __has_builtin(__builtin_amdgcn_s_wait_asynccnt)
#define HAVE_ASYNC_LDS 1
#endif
#endif

__device__ inline v8f wmma_f16(v16h a, v16h b, v8f c) {
  return __builtin_amdgcn_wmma_f32_16x16x32_f16(
      /*neg_a=*/false, a, /*neg_b=*/false, b,
      /*c_mod=*/(short)0, c, /*reuse_a=*/false, /*reuse_b=*/false);
}

// A fragment (16x32 f16): lane<16 -> row=lane&15, K = kc+{0..15};
// lane>=16 -> same rows, K = kc+{16..31}.
__device__ inline v16h a_frag_h(const _Float16* rowp, int kc, int hi) {
  v16h a;
  const _Float16* s0 = rowp + kc + hi * 16;
  const _Float16* s1 = s0 + 8;
#pragma unroll
  for (int i = 0; i < 8; ++i) { a[i] = s0[i]; a[8 + i] = s1[i]; }
  return a;
}

// B fragment (32x16 f16): lane holds 16 consecutive K values of one column;
// caller passes pointer to that contiguous run (already offset by hi*16).
__device__ inline v16h b_frag_h(const _Float16* colp) {
  v16h b;
#pragma unroll
  for (int i = 0; i < 16; ++i) b[i] = colp[i];
  return b;
}

// ------------------------- k0a: activation fp32->f16 ------------------------
__global__ __launch_bounds__(256) void xcvt_kernel(const float* __restrict__ X,
                                                   _Float16* __restrict__ Xh) {
  int idx = (blockIdx.x * 256 + threadIdx.x) * 4;
#pragma unroll
  for (int i = 0; i < 4; ++i) Xh[idx + i] = (_Float16)X[idx + i];
}

// --------------------------- k0b: weight transpose --------------------------
__global__ __launch_bounds__(256) void wtr_kernel(const float* __restrict__ W,
                                                  _Float16* __restrict__ WT) {
  int idx = blockIdx.x * 256 + threadIdx.x;     // 1M elements
  int n = idx >> 10, k = idx & 1023;
  WT[(size_t)n * Dq + k] = (_Float16)W[(size_t)k * Dq + n];
}

// ------------------ k1/k3: unified WMMA GEMM, 64x64 per wave ----------------
// mode 0: f16 out, [b][h][l][d] layout (K / Q projections)
// mode 1: f16 out, [b][h][d][l] layout (V projection, transposed)
// mode 2: fp32 out, plain [row][col]  (output projection)
__global__ __launch_bounds__(32) void gemm_kernel(
    const _Float16* __restrict__ A,   // [8192][1024] f16 row-major
    const _Float16* __restrict__ WT,  // [n][k] f16
    const float* __restrict__ bias,
    void* __restrict__ out, float scale, int mode) {
  int lane = threadIdx.x & 31, hi = lane >> 4, ln = lane & 15;
  int n0 = blockIdx.x * 64, m0 = blockIdx.y * 64;

  v8f acc[4][4] = {};                 // [row-block][col-block]
  const _Float16* arow[4];
#pragma unroll
  for (int rb = 0; rb < 4; ++rb)
    arow[rb] = A + (size_t)(m0 + rb * 16 + ln) * Dq;

  for (int kc = 0; kc < Dq; kc += 32) {
    v16h af[4];
#pragma unroll
    for (int rb = 0; rb < 4; ++rb) af[rb] = a_frag_h(arow[rb], kc, hi);
#pragma unroll
    for (int n = 0; n < 4; ++n) {
      const _Float16* colp =
          WT + (size_t)(n0 + n * 16 + ln) * Dq + kc + hi * 16;
      v16h bf = b_frag_h(colp);
#pragma unroll
      for (int rb = 0; rb < 4; ++rb)
        acc[rb][n] = wmma_f16(af[rb], bf, acc[rb][n]);   // 16 WMMA / chunk
    }
  }

#pragma unroll
  for (int n = 0; n < 4; ++n) {
    int col = n0 + n * 16 + ln;
    float bb = bias[col];
    int hh = col >> 6, dd = col & 63;
#pragma unroll
    for (int rb = 0; rb < 4; ++rb) {
#pragma unroll
      for (int r = 0; r < 8; ++r) {
        int row = m0 + rb * 16 + r + hi * 8;
        float v = (acc[rb][n][r] + bb) * scale;
        if (mode == 2) {
          ((float*)out)[(size_t)row * Dq + col] = v;
        } else {
          int bi = row >> 10, l = row & 1023;
          size_t oidx = (mode == 1)
              ? ((((size_t)bi * Hq + hh) * DHq + dd) * Lq + l)
              : ((((size_t)bi * Hq + hh) * Lq + l) * DHq + dd);
          ((_Float16*)out)[oidx] = (_Float16)v;
        }
      }
    }
  }
}

// --------------------------- k2: fused attention ----------------------------
// grid: (L/16, H, B); one wave32 per block, 16 q-rows, 1024-key strip in LDS.
__global__ __launch_bounds__(32) void attn_kernel(
    const _Float16* __restrict__ Qf, const _Float16* __restrict__ Kf,
    const _Float16* __restrict__ Vt, const float* __restrict__ zkg,
    const float* __restrict__ zvg, _Float16* __restrict__ ctxg,
    float* __restrict__ top) {
  __shared__ _Float16 lgh[16 * Lq];                       // 32 KB score strip
  __shared__ float qf[16 * DHq];                          // 4 KB
  __shared__ float rtab[16 * VOCABq];
  __shared__ __attribute__((aligned(16))) float zk_s[VOCABq * DHq];  // 8.25 KB
  __shared__ __attribute__((aligned(16))) float zv_s[VOCABq * DHq];  // 8.25 KB
  __shared__ float ctx_s[16 * DHq];                       // 4 KB
  __shared__ float redmax[2][16], redsum[2][16];

  int lane = threadIdx.x & 31, hi = lane >> 4, ln = lane & 15;
  int qt = blockIdx.x, h = blockIdx.y, b = blockIdx.z;
  int q0 = qt * 16;

  const _Float16* qg = Qf + (((size_t)b * Hq + h) * Lq + q0) * DHq;
  const _Float16* kg = Kf + (((size_t)b * Hq + h) * Lq) * DHq;
  const _Float16* vg = Vt + (((size_t)b * Hq + h) * DHq) * Lq;

#ifdef HAVE_ASYNC_LDS
  // CDNA5 async global->LDS copy for the relative tables (ASYNCcnt-tracked).
  for (int t = lane; t < (VOCABq * DHq) / 4; t += 32) {
    __builtin_amdgcn_global_load_async_to_lds_b128(
        (__attribute__((address_space(1))) v4i*)(zkg + t * 4),
        (__attribute__((address_space(3))) v4i*)(&zk_s[t * 4]), 0, 0);
    __builtin_amdgcn_global_load_async_to_lds_b128(
        (__attribute__((address_space(1))) v4i*)(zvg + t * 4),
        (__attribute__((address_space(3))) v4i*)(&zv_s[t * 4]), 0, 0);
  }
  __builtin_amdgcn_s_wait_asynccnt(0);
#else
  for (int t = lane; t < VOCABq * DHq; t += 32) {
    zk_s[t] = zkg[t];
    zv_s[t] = zvg[t];
  }
#endif
  for (int t = lane; t < 16 * DHq; t += 32) qf[t] = (float)qg[t];
  __syncthreads();

  // Per-row relative-logit table: rtab[row][j] = q_row . rel_k_table[j]
  for (int t = lane; t < 16 * VOCABq; t += 32) {
    int row = t / VOCABq, j = t % VOCABq;
    float acc = 0.f;
#pragma unroll 8
    for (int d = 0; d < DHq; ++d) acc += qf[row * DHq + d] * zk_s[j * DHq + d];
    rtab[t] = acc;
  }
  __syncthreads();

  // Q A-fragments (d-chunks 0..31 and 32..63)
  v16h a0 = a_frag_h(qg + (size_t)ln * DHq, 0, hi);
  v16h a1 = a_frag_h(qg + (size_t)ln * DHq, 32, hi);

  // Logits: 64 key-tiles of 16
  for (int kt = 0; kt < Lq / 16; ++kt) {
    int key = kt * 16 + ln;
    const _Float16* colp = kg + (size_t)key * DHq;
    if (kt + 1 < Lq / 16)
      __builtin_prefetch(colp + 16 * DHq, 0, 0);   // -> global_prefetch_b8
    v8f acc = {};
    acc = wmma_f16(a0, b_frag_h(colp + hi * 16), acc);
    acc = wmma_f16(a1, b_frag_h(colp + 32 + hi * 16), acc);
#pragma unroll
    for (int r = 0; r < 8; ++r) {
      int row = r + hi * 8;
      int dist = (q0 + row) - key;
      dist = dist < -16 ? -16 : (dist > 16 ? 16 : dist);
      lgh[row * Lq + key] = (_Float16)(acc[r] + rtab[row * VOCABq + dist + 16]);
    }
  }
  __syncthreads();

  // Softmax: lane pair (ln, ln+16) shares row ln, each scans 512 keys.
  {
    int kb = hi * 512;
    float m = -1e30f;
    for (int k = 0; k < 512; ++k) {
      float v = (float)lgh[ln * Lq + kb + k];
      m = fmaxf(m, v);
    }
    redmax[hi][ln] = m;
    __syncthreads();
    float M = fmaxf(redmax[0][ln], redmax[1][ln]);
    float s = 0.f;
    for (int k = 0; k < 512; ++k) {
      float w = __expf((float)lgh[ln * Lq + kb + k] - M);
      lgh[ln * Lq + kb + k] = (_Float16)w;
      s += w;
    }
    redsum[hi][ln] = s;
    __syncthreads();
    float inv = 1.f / (redsum[0][ln] + redsum[1][ln]);
    float* ta = top + ((size_t)b * Lq + (q0 + ln)) * Lq;
    for (int k = 0; k < 512; ++k) {
      float w = (float)lgh[ln * Lq + kb + k] * inv;
      lgh[ln * Lq + kb + k] = (_Float16)w;
      if (h == 0) ta[kb + k] = w;           // top_attn = weights[:, 0]
    }
  }
  __syncthreads();

  // ctx = P @ V  (A = weights f16 from LDS, B = V^T contiguous per dim)
  v8f cacc[4] = {};
  for (int kc = 0; kc < Lq; kc += 32) {
    v16h a = a_frag_h(&lgh[ln * Lq], kc, hi);
#pragma unroll
    for (int n = 0; n < 4; ++n) {
      const _Float16* colp = vg + (size_t)(n * 16 + ln) * Lq + kc + hi * 16;
      cacc[n] = wmma_f16(a, b_frag_h(colp), cacc[n]);
    }
  }
#pragma unroll
  for (int n = 0; n < 4; ++n)
#pragma unroll
    for (int r = 0; r < 8; ++r)
      ctx_s[(r + hi * 8) * DHq + n * 16 + ln] = cacc[n][r];
  __syncthreads();

  // Relative-value term via 33 bucket sums; softmax rows sum to 1.
  if (lane < 16) {
    int qg_row = q0 + ln;
    float s[VOCABq];
    float p = 0.f;
    int kmax = qg_row - 16;                  // bucket 32: k <= q-16
    for (int k = 0; k <= kmax; ++k) p += (float)lgh[ln * Lq + k];
    s[32] = p;
    float tot = p;
#pragma unroll
    for (int j = 1; j < 32; ++j) {           // buckets 1..31: k = q+16-j
      int k = qg_row + 16 - j;
      float w = (k >= 0 && k < Lq) ? (float)lgh[ln * Lq + k] : 0.f;
      s[j] = w;
      tot += w;
    }
    s[0] = 1.f - tot;                        // bucket 0: k >= q+16
    for (int d = 0; d < DHq; ++d) {
      float acc = 0.f;
#pragma unroll
      for (int j = 0; j < VOCABq; ++j) acc += s[j] * zv_s[j * DHq + d];
      ctx_s[ln * DHq + d] += acc;
    }
  }
  __syncthreads();

  // ctx -> f16 workspace [b][l][h*64+d]
  for (int t = lane; t < 16 * DHq; t += 32) {
    int row = t / DHq, d = t % DHq;
    ctxg[((size_t)b * Lq + q0 + row) * Dq + h * DHq + d] = (_Float16)ctx_s[t];
  }
}

// ----------------------------------------------------------------------------
extern "C" void kernel_launch(void* const* d_in, const int* in_sizes, int n_in,
                              void* d_out, int out_size, void* d_ws, size_t ws_size,
                              hipStream_t stream) {
  const float* key   = (const float*)d_in[0];
  const float* value = (const float*)d_in[1];
  const float* query = (const float*)d_in[2];
  const float* Wk = (const float*)d_in[3];
  const float* bk = (const float*)d_in[4];
  const float* Wv = (const float*)d_in[5];
  const float* bv = (const float*)d_in[6];
  const float* Wq = (const float*)d_in[7];
  const float* bq = (const float*)d_in[8];
  const float* Wo = (const float*)d_in[9];
  const float* bo = (const float*)d_in[10];
  const float* rel_k = (const float*)d_in[11];
  const float* rel_v = (const float*)d_in[12];

  char* ws = (char*)d_ws;
  const size_t MB = 1u << 20;
  _Float16* WkT = (_Float16*)(ws + 0 * MB);
  _Float16* WvT = (_Float16*)(ws + 2 * MB);
  _Float16* WqT = (_Float16*)(ws + 4 * MB);
  _Float16* WoT = (_Float16*)(ws + 6 * MB);
  _Float16* Kf  = (_Float16*)(ws + 8 * MB);    // [b][h][l][d]
  _Float16* Vt  = (_Float16*)(ws + 24 * MB);   // [b][h][d][l]
  _Float16* Qf  = (_Float16*)(ws + 40 * MB);   // [b][h][l][d], pre-scaled
  _Float16* ctx = (_Float16*)(ws + 56 * MB);   // [b*l][D]
  _Float16* Kx  = (_Float16*)(ws + 72 * MB);   // key   f16
  _Float16* Vx  = (_Float16*)(ws + 88 * MB);   // value f16
  _Float16* Qx  = (_Float16*)(ws + 104 * MB);  // query f16

  float* out_p = (float*)d_out;
  float* top_p = out_p + (size_t)Bq * Lq * Dq;

  // k0a: activations fp32 -> f16
  xcvt_kernel<<<8192, 256, 0, stream>>>(key, Kx);
  xcvt_kernel<<<8192, 256, 0, stream>>>(value, Vx);
  xcvt_kernel<<<8192, 256, 0, stream>>>(query, Qx);

  // k0b: weight transposes (fp32 [k][n] -> f16 [n][k])
  wtr_kernel<<<4096, 256, 0, stream>>>(Wk, WkT);
  wtr_kernel<<<4096, 256, 0, stream>>>(Wv, WvT);
  wtr_kernel<<<4096, 256, 0, stream>>>(Wq, WqT);
  wtr_kernel<<<4096, 256, 0, stream>>>(Wo, WoT);

  // k1: projections (M = B*L = 8192 rows, N = 1024), 64x64 tile per wave
  dim3 gg(Dq / 64, (Bq * Lq) / 64);
  gemm_kernel<<<gg, 32, 0, stream>>>(Kx, WkT, bk, Kf, 1.0f,   0);
  gemm_kernel<<<gg, 32, 0, stream>>>(Vx, WvT, bv, Vt, 1.0f,   1);
  gemm_kernel<<<gg, 32, 0, stream>>>(Qx, WqT, bq, Qf, 0.125f, 0); // /sqrt(64)

  // k2: fused attention
  attn_kernel<<<dim3(Lq / 16, Hq, Bq), 32, 0, stream>>>(Qf, Kf, Vt, rel_k,
                                                        rel_v, ctx, top_p);

  // k3: output projection (fp32 out + bias)
  gemm_kernel<<<gg, 32, 0, stream>>>(ctx, WoT, bo, out_p, 1.0f, 2);
}
